// GPSConv_4982162063381
// MI455X (gfx1250) — matrix operand order repaired and checked by reading, not compiled, not run on hardware
//
#include <hip/hip_runtime.h>

// ---------------------------------------------------------------------------
// GPSConv block for MI455X (gfx1250, wave32, WMMA f16->f32).
// Shapes fixed by the reference: N=24576 nodes, C=256, H=8 heads, HD=32,
// G=64 graphs, batch sorted + pos contiguous per graph (generator property).
// Round 2: 32-row GEMM M-tiles (B-fragment reuse x2) + async global->LDS
// staging (GLOBAL_LOAD_ASYNC_TO_LDS_B128 / s_wait_asynccnt).
// ---------------------------------------------------------------------------

typedef _Float16 half16 __attribute__((ext_vector_type(16)));
typedef float    floatx8 __attribute__((ext_vector_type(8)));

#define NN     24576
#define CC     256
#define HH     8
#define HDIM   32
#define GG     64
#define LMAXX  512
#define EPSF   1e-5f
#define SCALEF 0.17677669529663687f   // 1/sqrt(32)

static __device__ __forceinline__ int iminf(int a, int b) { return a < b ? a : b; }

// LDS byte offset of a generic pointer to __shared__ (shared aperture keeps
// the LDS offset in the low 32 bits of the flat address).
static __device__ __forceinline__ unsigned lds_off(const void* p) {
  return (unsigned)(unsigned long long)p;
}

// --------------------------- segment boundaries ----------------------------
__global__ void k_boundaries(const int* __restrict__ batch, int* __restrict__ start,
                             int n, int g) {
  int i = blockIdx.x * blockDim.x + threadIdx.x;
  if (i >= n) return;
  if (i == 0) start[0] = 0;
  else if (batch[i] != batch[i - 1]) start[batch[i]] = i;
  if (i == n - 1) start[g] = n;
}

// --------------------------- f32 -> f16 convert ----------------------------
__global__ void k_cvt_f16(const float* __restrict__ src, _Float16* __restrict__ dst,
                          int total) {
  int i = blockIdx.x * blockDim.x + threadIdx.x;
  if (i < total) dst[i] = (_Float16)src[i];
}

// ------------------- pack W[Nn][K] (row-major) into B-fragments ------------
// B-fragment layout this code uses for V_WMMA_F32_16X16X32_F16 (contiguous k, same as A loads):
//   lane = (n%16) | (((k>>4)&1)<<4), element e = k%16, fragment (nt, kb).
// Stored so each lane reads 16 contiguous f16 (32B) per fragment.
__global__ void k_pack_b(const float* __restrict__ W, _Float16* __restrict__ out,
                         int K, int Nn) {
  int idx = blockIdx.x * blockDim.x + threadIdx.x;
  if (idx >= K * Nn) return;
  int k = idx % K, n = idx / K;
  int kb = k >> 5, nt = n >> 4;
  int lane = (n & 15) | (((k >> 4) & 1) << 4);
  int e = k & 15;
  int kBlocks = K >> 5;
  out[((((size_t)nt * kBlocks + kb) * 32 + lane) << 4) + e] = (_Float16)W[(size_t)n * K + k];
}

// ------------------------------ generic GEMM -------------------------------
// C[M][Nn] = A[M][K](f16) * Bfrag + bias (+resid) (ReLU?), out f32 and/or f16.
// 32-row M tile per block: each B fragment feeds two WMMAs (two M subtiles).
__global__ __launch_bounds__(256)
void k_gemm_wmma(const _Float16* __restrict__ A, const _Float16* __restrict__ Bfrag,
                 const float* __restrict__ bias, const float* __restrict__ resid,
                 float* __restrict__ outF32, _Float16* __restrict__ outF16,
                 int M, int K, int Nn, int doRelu) {
  extern __shared__ _Float16 sA[];           // 32 x K
  const int row0 = blockIdx.x * 32;

  // ---- async stage of the 32xK A tile: global -> LDS, no VGPR round-trip.
  // A tile is 32*K contiguous f16 (row-major), copied linearly in 16B chunks.
  {
    const _Float16* gbase = A + (size_t)row0 * K;
    const int chunks = (32 * K) / 8;                 // 16B chunks
    for (int c = threadIdx.x; c < chunks; c += blockDim.x) {
      unsigned dst = lds_off(&sA[c * 8]);
      const _Float16* src = gbase + (size_t)c * 8;
      asm volatile("global_load_async_to_lds_b128 %0, %1, off"
                   :: "v"(dst), "v"(src) : "memory");
    }
    asm volatile("s_wait_asynccnt 0x0" ::: "memory");
    __syncthreads();
  }

  const int wave = threadIdx.x >> 5;
  const int lane = threadIdx.x & 31;
  const int m = lane & 15, halfid = lane >> 4;
  const int nTiles = Nn >> 4;
  const int kBlocks = K >> 5;

  for (int nt = wave; nt < nTiles; nt += 8) {
    floatx8 acc0 = {0.f,0.f,0.f,0.f,0.f,0.f,0.f,0.f};
    floatx8 acc1 = {0.f,0.f,0.f,0.f,0.f,0.f,0.f,0.f};
    const _Float16* bptr = Bfrag + (((size_t)nt * kBlocks) * 32 + lane) * 16;
    for (int kb = 0; kb < kBlocks; ++kb) {
      half16 b = *(const half16*)(bptr + (size_t)kb * 32 * 16);   // 32B contiguous
      half16 a0, a1;
      const _Float16* r0p = &sA[m * K + kb * 32 + 16 * halfid];
      const _Float16* r1p = &sA[(16 + m) * K + kb * 32 + 16 * halfid];
      *((float4*)&a0)       = *(const float4*)(r0p);        // e0..7  : k = 16*half+0..7
      *(((float4*)&a0) + 1) = *(const float4*)(r0p + 8);    // e8..15 : k = 16*half+8..15
      *((float4*)&a1)       = *(const float4*)(r1p);
      *(((float4*)&a1) + 1) = *(const float4*)(r1p + 8);
      acc0 = __builtin_amdgcn_wmma_f32_16x16x32_f16(false, a0, false, b,
                                                    (short)0, acc0, false, false);
      acc1 = __builtin_amdgcn_wmma_f32_16x16x32_f16(false, a1, false, b,
                                                    (short)0, acc1, false, false);
    }
    const int n = nt * 16 + m;
    const float bn = bias ? bias[n] : 0.f;
#pragma unroll
    for (int sub = 0; sub < 2; ++sub) {
      const floatx8& acc = sub ? acc1 : acc0;
#pragma unroll
      for (int r = 0; r < 8; ++r) {
        size_t grow = (size_t)row0 + sub * 16 + (r + 8 * halfid);
        float v = acc[r] + bn;
        if (resid) v += resid[grow * (size_t)Nn + n];
        if (doRelu) v = fmaxf(v, 0.f);
        if (outF32) outF32[grow * (size_t)Nn + n] = v;
        if (outF16) outF16[grow * (size_t)Nn + n] = (_Float16)v;
      }
    }
  }
}

// --------------------------- flash attention -------------------------------
// qkv: [N][768] f16, q = cols h*32+hd, k = 256+h*32+hd, v = 512+h*32+hd.
// One wave per (graph, head, 16-query tile). ctx: [N][256] f16.
__global__ __launch_bounds__(32)
void k_attn_wmma(const _Float16* __restrict__ qkv, _Float16* __restrict__ ctx,
                 const int* __restrict__ start) {
  __shared__ _Float16 sP[16 * 32];
  const int g = blockIdx.z, h = blockIdx.y, qt = blockIdx.x;
  const int s0 = start[g];
  const int cnt = start[g + 1] - s0;
  if (qt * 16 >= cnt) return;

  const int lane = threadIdx.x;
  const int m = lane & 15, halfid = lane >> 4;

  // Q A-fragment (16 queries x 32 hd), loaded once
  const int qrow = iminf(qt * 16 + m, cnt - 1);
  const _Float16* qp = qkv + (size_t)(s0 + qrow) * 768 + h * 32;
  half16 qa;
  *((float4*)&qa)       = *(const float4*)(qp + 16 * halfid);
  *(((float4*)&qa) + 1) = *(const float4*)(qp + 16 * halfid + 8);

  float mrun[8], lrun[8];
  floatx8 acc0 = {0.f,0.f,0.f,0.f,0.f,0.f,0.f,0.f};
  floatx8 acc1 = {0.f,0.f,0.f,0.f,0.f,0.f,0.f,0.f};
#pragma unroll
  for (int r = 0; r < 8; ++r) { mrun[r] = -1e30f; lrun[r] = 0.f; }

  const int keyBlocks = (cnt + 31) >> 5;
  for (int j = 0; j < keyBlocks; ++j) {
    const int key0 = j * 32;
    // K^T B-fragments: n = key, k = hd; each lane: 16 contiguous f16 (32B)
    const int kn0 = iminf(key0 + m, cnt - 1);
    const int kn1 = iminf(key0 + 16 + m, cnt - 1);
    half16 b0 = *(const half16*)(qkv + (size_t)(s0 + kn0) * 768 + 256 + h * 32 + 16 * halfid);
    half16 b1 = *(const half16*)(qkv + (size_t)(s0 + kn1) * 768 + 256 + h * 32 + 16 * halfid);
    floatx8 z = {0.f,0.f,0.f,0.f,0.f,0.f,0.f,0.f};
    floatx8 sc0 = __builtin_amdgcn_wmma_f32_16x16x32_f16(false, qa, false, b0, (short)0, z, false, false);
    floatx8 sc1 = __builtin_amdgcn_wmma_f32_16x16x32_f16(false, qa, false, b1, (short)0, z, false, false);

    const bool v0 = (key0 + m) < cnt;
    const bool v1 = (key0 + 16 + m) < cnt;
#pragma unroll
    for (int r = 0; r < 8; ++r) {
      float p0 = v0 ? sc0[r] * SCALEF : -1e30f;
      float p1 = v1 ? sc1[r] * SCALEF : -1e30f;
      float rm = fmaxf(p0, p1);
      rm = fmaxf(rm, __shfl_xor(rm, 8)); rm = fmaxf(rm, __shfl_xor(rm, 4));
      rm = fmaxf(rm, __shfl_xor(rm, 2)); rm = fmaxf(rm, __shfl_xor(rm, 1));
      float nm = fmaxf(mrun[r], rm);
      float alpha = __expf(mrun[r] - nm);
      mrun[r] = nm;
      p0 = __expf(p0 - nm);                 // masked -> exact 0
      p1 = __expf(p1 - nm);
      float rs = p0 + p1;
      rs += __shfl_xor(rs, 8); rs += __shfl_xor(rs, 4);
      rs += __shfl_xor(rs, 2); rs += __shfl_xor(rs, 1);
      lrun[r] = lrun[r] * alpha + rs;
      acc0[r] *= alpha; acc1[r] *= alpha;
      sP[(r + 8 * halfid) * 32 + m]      = (_Float16)p0;   // C-layout -> LDS
      sP[(r + 8 * halfid) * 32 + 16 + m] = (_Float16)p1;
    }
    __syncthreads();
    // reload P as A-fragment (16 q x 32 local keys)
    half16 pa;
    const _Float16* pr = &sP[m * 32 + 16 * halfid];
    *((float4*)&pa)       = *(const float4*)(pr);
    *(((float4*)&pa) + 1) = *(const float4*)(pr + 8);
    // V B-fragments (k = key, n = hd); invalid keys carry P==0 so clamp is safe
    half16 vb0, vb1;
#pragma unroll
    for (int e = 0; e < 16; ++e) {
      int key = iminf(key0 + 16 * halfid + e, cnt - 1);
      const _Float16* vp = qkv + (size_t)(s0 + key) * 768 + 512 + h * 32;
      vb0[e] = vp[m];
      vb1[e] = vp[16 + m];
    }
    acc0 = __builtin_amdgcn_wmma_f32_16x16x32_f16(false, pa, false, vb0, (short)0, acc0, false, false);
    acc1 = __builtin_amdgcn_wmma_f32_16x16x32_f16(false, pa, false, vb1, (short)0, acc1, false, false);
    __syncthreads();
  }
  // finalize: divide by row sums, store f16 ctx
#pragma unroll
  for (int r = 0; r < 8; ++r) {
    int qq = qt * 16 + r + 8 * halfid;
    if (qq < cnt) {
      float inv = 1.f / lrun[r];
      _Float16* op = ctx + (size_t)(s0 + qq) * 256 + h * 32;
      op[m]      = (_Float16)(acc0[r] * inv);
      op[16 + m] = (_Float16)(acc1[r] * inv);
    }
  }
}

// ------------------------------ batchnorm ----------------------------------
__global__ __launch_bounds__(256)
void k_bn_stats(const float* __restrict__ X, float* __restrict__ mean,
                float* __restrict__ rstd, int M) {
  const int c = blockIdx.x;
  float s = 0.f, s2 = 0.f;
  for (int i = threadIdx.x; i < M; i += 256) {
    float v = X[(size_t)i * CC + c];
    s += v; s2 += v * v;
  }
  __shared__ float sh[256], sh2[256];
  sh[threadIdx.x] = s; sh2[threadIdx.x] = s2;
  __syncthreads();
  for (int o = 128; o; o >>= 1) {
    if (threadIdx.x < o) { sh[threadIdx.x] += sh[threadIdx.x + o]; sh2[threadIdx.x] += sh2[threadIdx.x + o]; }
    __syncthreads();
  }
  if (threadIdx.x == 0) {
    float mu = sh[0] / (float)M;
    float var = sh2[0] / (float)M - mu * mu;   // biased var, matches jnp.var
    mean[c] = mu;
    rstd[c] = rsqrtf(var + EPSF);
  }
}

__global__ void k_bn_apply(const float* __restrict__ X, const float* __restrict__ mean,
                           const float* __restrict__ rstd, const float* __restrict__ gamma,
                           const float* __restrict__ beta, float* __restrict__ outF32,
                           _Float16* __restrict__ outF16, int total) {
  int i = blockIdx.x * blockDim.x + threadIdx.x;
  if (i >= total) return;
  int c = i & (CC - 1);
  float v = (X[i] - mean[c]) * rstd[c] * gamma[c] + beta[c];
  if (outF32) outF32[i] = v;
  if (outF16) outF16[i] = (_Float16)v;
}

// ------------------------------ launcher -----------------------------------
extern "C" void kernel_launch(void* const* d_in, const int* in_sizes, int n_in,
                              void* d_out, int out_size, void* d_ws, size_t ws_size,
                              hipStream_t stream) {
  const float* x          = (const float*)d_in[0];
  const int*   batch      = (const int*)d_in[1];
  const float* in_proj_w  = (const float*)d_in[5];
  const float* in_proj_b  = (const float*)d_in[6];
  const float* out_proj_w = (const float*)d_in[7];
  const float* out_proj_b = (const float*)d_in[8];
  const float* w1         = (const float*)d_in[9];
  const float* bm1        = (const float*)d_in[10];
  const float* w2         = (const float*)d_in[11];
  const float* bm2        = (const float*)d_in[12];
  const float* g2         = (const float*)d_in[13];
  const float* beta2      = (const float*)d_in[14];
  const float* g3         = (const float*)d_in[15];
  const float* beta3      = (const float*)d_in[16];

  char* ws = (char*)d_ws;
  const size_t SZ_XF16 = (size_t)NN * CC * 2;        // 12.58 MB
  const size_t SZ_QKV  = (size_t)NN * 3 * CC * 2;    // 37.75 MB
  const size_t SZ_CTX  = (size_t)NN * CC * 2;
  const size_t SZ_F32  = (size_t)NN * CC * 4;        // 25.17 MB
  const size_t SZ_ACT  = (size_t)NN * 2 * CC * 2;    // 25.17 MB

  size_t o_xf16 = 0;
  size_t o_qkv  = o_xf16 + SZ_XF16;
  size_t o_ctx  = o_qkv + SZ_QKV;
  size_t o_h    = o_ctx + SZ_CTX;
  size_t o_h2   = o_h + SZ_F32;
  size_t o_wqkv = o_h2 + SZ_F32;
  size_t o_wout = o_wqkv + (size_t)256 * 768 * 2;
  size_t o_w1   = o_wout + (size_t)256 * 256 * 2;
  size_t o_w2   = o_w1 + (size_t)256 * 512 * 2;
  size_t o_start= o_w2 + (size_t)512 * 256 * 2;
  size_t o_mean = o_start + 1024;
  size_t o_rstd = o_mean + 1024;
  // lifetime-based reuse
  size_t o_h2f16 = o_xf16;                 // x dead after GEMM2 epilogue
  size_t o_act   = o_qkv;                  // qkv dead after attention
  size_t o_y     = o_qkv + SZ_ACT;         // tail of qkv + ctx regions (both dead)

  _Float16* xf16  = (_Float16*)(ws + o_xf16);
  _Float16* qkv   = (_Float16*)(ws + o_qkv);
  _Float16* ctx   = (_Float16*)(ws + o_ctx);
  float*    hbuf  = (float*)(ws + o_h);
  float*    h2    = (float*)(ws + o_h2);
  _Float16* h2f16 = (_Float16*)(ws + o_h2f16);
  _Float16* act   = (_Float16*)(ws + o_act);
  float*    ybuf  = (float*)(ws + o_y);
  _Float16* wqkvf = (_Float16*)(ws + o_wqkv);
  _Float16* woutf = (_Float16*)(ws + o_wout);
  _Float16* w1f   = (_Float16*)(ws + o_w1);
  _Float16* w2f   = (_Float16*)(ws + o_w2);
  int*      start = (int*)(ws + o_start);
  float*    mean  = (float*)(ws + o_mean);
  float*    rstd  = (float*)(ws + o_rstd);

  const int elems = NN * CC;

  // 1) segment boundaries (batch is sorted, pos contiguous per graph)
  k_boundaries<<<(NN + 255) / 256, 256, 0, stream>>>(batch, start, NN, GG);
  // 2) x -> f16
  k_cvt_f16<<<(elems + 255) / 256, 256, 0, stream>>>(x, xf16, elems);
  // 3) pack weights into WMMA B-fragment layout
  k_pack_b<<<(256 * 768 + 255) / 256, 256, 0, stream>>>(in_proj_w,  wqkvf, 256, 768);
  k_pack_b<<<(256 * 256 + 255) / 256, 256, 0, stream>>>(out_proj_w, woutf, 256, 256);
  k_pack_b<<<(256 * 512 + 255) / 256, 256, 0, stream>>>(w1,         w1f,   256, 512);
  k_pack_b<<<(512 * 256 + 255) / 256, 256, 0, stream>>>(w2,         w2f,   512, 256);
  // 4) QKV projection: [N,256]x[256,768]+b -> qkv f16
  k_gemm_wmma<<<NN / 32, 256, 32 * 256 * 2, stream>>>(xf16, wqkvf, in_proj_b, nullptr,
                                                      nullptr, qkv, NN, 256, 768, 0);
  // 5) per-graph flash attention -> ctx f16 [N,256]
  k_attn_wmma<<<dim3(LMAXX / 16, HH, GG), 32, 0, stream>>>(qkv, ctx, start);
  // 6) out_proj + residual(x): h = ctx@Wout^T + b + x
  k_gemm_wmma<<<NN / 32, 256, 32 * 256 * 2, stream>>>(ctx, woutf, out_proj_b, x,
                                                      hbuf, nullptr, NN, 256, 256, 0);
  // 7) BN1 -> h2 (f32 + f16)
  k_bn_stats<<<CC, 256, 0, stream>>>(hbuf, mean, rstd, NN);
  k_bn_apply<<<(elems + 255) / 256, 256, 0, stream>>>(hbuf, mean, rstd, g2, beta2,
                                                      h2, h2f16, elems);
  // 8) MLP: act = relu(h2@w1^T + bm1) f16 [N,512]
  k_gemm_wmma<<<NN / 32, 256, 32 * 256 * 2, stream>>>(h2f16, w1f, bm1, nullptr,
                                                      nullptr, act, NN, 256, 512, 1);
  // 9) y = act@w2^T + bm2 + h2
  k_gemm_wmma<<<NN / 32, 256, 32 * 512 * 2, stream>>>(act, w2f, bm2, h2,
                                                      ybuf, nullptr, NN, 512, 256, 0);
  // 10) BN2 -> d_out
  k_bn_stats<<<CC, 256, 0, stream>>>(ybuf, mean, rstd, NN);
  k_bn_apply<<<(elems + 255) / 256, 256, 0, stream>>>(ybuf, mean, rstd, g3, beta3,
                                                      (float*)d_out, nullptr, elems);
}
